// MSDeformAttn_64750926954661
// MI455X (gfx1250) — compile-verified, matrix operand order checked by reading, not run on hardware
//
#include <hip/hip_runtime.h>

// ---------------------------------------------------------------------------
// MS-Deformable Attention for MI455X (gfx1250, wave32, WMMA).
//
// Pipeline (all on `stream`):
//   0. transpose+convert the 4 weight matrices to f16 N x K   (tiny)
//   1. value = input_flatten @ W_val + b_val   (WMMA f16 GEMM, LDS-staged A)
//   2. off   = query @ W_off + b_off           (WMMA f16 GEMM)
//   3. attnl = query @ W_attn + b_attn         (WMMA f16 GEMM)
//   4. softmax + bilinear sampling + weighted accumulation
//   5. out   = acc @ W_out + b_out             (WMMA f16 GEMM -> d_out)
//
// GEMM: block = 256 thr (8 waves), block tile 32 x N, waves 2(M) x 4(N),
// each wave holds NT=N/64 fp32 accumulators, reusing one A fragment per
// k-step across NT v_wmma_f32_16x16x32_f16 ops. A tile staged to LDS as f16
// (stride 264 f16 => conflict-free b128 fragment reads); B fragments are two
// contiguous global_load_b128 from the pre-transposed f16 weights.
// ---------------------------------------------------------------------------

typedef __attribute__((ext_vector_type(16))) _Float16 v16h;
typedef __attribute__((ext_vector_type(8)))  _Float16 v8h;
typedef __attribute__((ext_vector_type(8)))  float    v8f;

#define D_MODEL   256
#define N_HEADS   8
#define HEAD_DIM  32
#define N_LEVELS  4
#define N_POINTS  4
#define LEN_TOTAL 20197            // 100*152 + 50*76 + 25*38 + 13*19
#define NBATCH    2
#define M_ROWS    (NBATCH * LEN_TOTAL)   // 40394
#define LDS_STRIDE 264             // 256 + 8 f16 pad: fragment reads hit 64 banks

// ---------------------------------------------------------------------------
// Weight transform: W (K x N, f32, row-major) -> Wt (N x K, f16, row-major).
// One-time, 65K elements max; strided reads are irrelevant at this size.
// ---------------------------------------------------------------------------
template <int N>
__global__ __launch_bounds__(256)
void transpose_w_f16(const float* __restrict__ W, _Float16* __restrict__ Wt) {
  const int idx = blockIdx.x * 256 + threadIdx.x;   // over N*256
  const int n = idx >> 8;
  const int k = idx & 255;
  Wt[idx] = (_Float16)W[(size_t)k * N + n];
}

// ---------------------------------------------------------------------------
// WMMA GEMM: C[M x N] = f16(A[M x 256]) @ f16(Wt^T) + bias, fp32 accumulate.
// grid.x = ceil(M/32). Fragment layouts per CDNA5 ISA 7.12.2 (wave32):
//   A 16x32 f16 : lane-half 0: row=lane,    K = [k0,k0+8) u [k0+16,k0+24)
//                 lane-half 1: row=lane-16, K shifted by +8
//   B 32x16 f16 : mirrored, lane selects column N = lane&15
//   C/D 16x16 f32: vgpr v holds M = v + 8*(lane>=16), N = lane&15
// ---------------------------------------------------------------------------
template <int N, int NT>   // NT = N/64 accumulators per wave
__global__ __launch_bounds__(256)
void wmma_gemm_bias(const float* __restrict__ A,
                    const _Float16* __restrict__ Wt,   // N x 256 f16
                    const float* __restrict__ bias,
                    float* __restrict__ C, int M) {
  constexpr int K = 256;
  __shared__ _Float16 As[32 * LDS_STRIDE];

  const int tid = threadIdx.x;
  const int m0  = blockIdx.x * 32;

  // ---- stage A block (32 x 256) into LDS as f16, one pass, no redundancy ----
  {
    const int r   = tid >> 3;             // 0..31
    const int cb  = (tid & 7) * 32;       // 0..224
    const int row = min(m0 + r, M - 1);   // clamp keeps loads in-bounds
    const float* src = A + (size_t)row * K + cb;
    _Float16*    dst = &As[r * LDS_STRIDE + cb];
#pragma unroll
    for (int i = 0; i < 4; ++i) {
      float4 f0 = *(const float4*)(src + i * 8);
      float4 f1 = *(const float4*)(src + i * 8 + 4);
      v8h h;
      h[0] = (_Float16)f0.x; h[1] = (_Float16)f0.y;
      h[2] = (_Float16)f0.z; h[3] = (_Float16)f0.w;
      h[4] = (_Float16)f1.x; h[5] = (_Float16)f1.y;
      h[6] = (_Float16)f1.z; h[7] = (_Float16)f1.w;
      *(v8h*)(dst + i * 8) = h;
    }
  }
  __syncthreads();

  const int wave = tid >> 5;
  const int lane = tid & 31;
  const int half = lane >> 4;            // 0 or 1
  const int l16  = lane & 15;
  const int wm   = wave >> 2;            // 0..1 : M sub-tile
  const int wn   = wave & 3;             // 0..3 : N quarter
  const int mrow = wm * 16 + l16;        // LDS row for A fragment
  const int nb   = wn * (N / 4);         // global col base for this wave

  v8f c[NT];
#pragma unroll
  for (int t = 0; t < NT; ++t) c[t] = (v8f){};

#pragma unroll
  for (int k0 = 0; k0 < K; k0 += 32) {
    // A fragment: two 16B LDS reads, shared across all NT WMMAs
    const _Float16* ap = &As[mrow * LDS_STRIDE + k0 + half * 8];
    v8h a_lo = *(const v8h*)(ap);
    v8h a_hi = *(const v8h*)(ap + 16);
    v16h a;
#pragma unroll
    for (int i = 0; i < 8; ++i) { a[i] = a_lo[i]; a[i + 8] = a_hi[i]; }

#pragma unroll
    for (int t = 0; t < NT; ++t) {
      // B fragment: two contiguous global_load_b128 from f16 N x K weights
      const _Float16* bp = Wt + (size_t)(nb + t * 16 + l16) * K + k0 + half * 8;
      v8h b_lo = *(const v8h*)(bp);
      v8h b_hi = *(const v8h*)(bp + 16);
      v16h b;
#pragma unroll
      for (int i = 0; i < 8; ++i) { b[i] = b_lo[i]; b[i + 8] = b_hi[i]; }

      c[t] = __builtin_amdgcn_wmma_f32_16x16x32_f16(false, a, false, b,
                                                    (short)0, c[t], false, false);
    }
  }

#pragma unroll
  for (int t = 0; t < NT; ++t) {
    const int   col = nb + t * 16 + l16;
    const float bv  = bias[col];
#pragma unroll
    for (int v = 0; v < 8; ++v) {
      const int r = m0 + wm * 16 + half * 8 + v;
      if (r < M) C[(size_t)r * N + col] = c[t][v] + bv;
    }
  }
}

// ---------------------------------------------------------------------------
// Softmax + bilinear sampling + attention-weighted accumulation.
// grid = (LEN_TOTAL, NBATCH), block = 256: wave = head, lane = channel (d=32).
// Each tap is one 128B coalesced gather; scalar math replicated per lane.
// ---------------------------------------------------------------------------
__global__ __launch_bounds__(256)
void msda_sample(const float* __restrict__ refpts,   // (N, Len, 4, 2)
                 const float* __restrict__ value,    // (N*Len, 256)
                 const float* __restrict__ off,      // (N*Len, 256)
                 const float* __restrict__ logits,   // (N*Len, 128)
                 float* __restrict__ acc) {          // (N*Len, 256)
  constexpr int LVL_H[N_LEVELS] = {100, 50, 25, 13};
  constexpr int LVL_W[N_LEVELS] = {152, 76, 38, 19};
  constexpr int LVL_S[N_LEVELS] = {0, 15200, 19000, 19950};

  const int q    = blockIdx.x;
  const int n    = blockIdx.y;
  const int h    = threadIdx.x >> 5;
  const int lane = threadIdx.x & 31;
  const size_t qf = (size_t)n * LEN_TOTAL + q;

  const float* lg = logits + qf * (N_HEADS * N_LEVELS * N_POINTS)
                           + h * (N_LEVELS * N_POINTS);
  float w[16];
  float mx = -3.0e38f;
#pragma unroll
  for (int i = 0; i < 16; ++i) { w[i] = lg[i]; mx = fmaxf(mx, w[i]); }
  float s = 0.f;
#pragma unroll
  for (int i = 0; i < 16; ++i) { w[i] = __expf(w[i] - mx); s += w[i]; }
  const float inv = 1.f / s;

  const float* ofs = off + qf * D_MODEL + h * (N_LEVELS * N_POINTS * 2);
  const float* rp  = refpts + qf * (N_LEVELS * 2);

  float accf = 0.f;
#pragma unroll
  for (int l = 0; l < N_LEVELS; ++l) {
    const int   H  = LVL_H[l], W = LVL_W[l];
    const float rx = rp[l * 2 + 0];
    const float ry = rp[l * 2 + 1];
    const float* vb = value + ((size_t)n * LEN_TOTAL + LVL_S[l]) * D_MODEL
                            + h * HEAD_DIM + lane;
#pragma unroll
    for (int p = 0; p < N_POINTS; ++p) {
      const float ox = ofs[(l * N_POINTS + p) * 2 + 0];
      const float oy = ofs[(l * N_POINTS + p) * 2 + 1];
      const float x  = (rx + ox / (float)W) * (float)W - 0.5f;
      const float y  = (ry + oy / (float)H) * (float)H - 0.5f;
      const float x0f = floorf(x), y0f = floorf(y);
      const float fx = x - x0f, fy = y - y0f;
      const int   x0 = (int)x0f, y0 = (int)y0f;
      const float aw = w[l * N_POINTS + p] * inv;
      const float tw[4] = {(1.f - fx) * (1.f - fy), fx * (1.f - fy),
                           (1.f - fx) * fy,          fx * fy};
#pragma unroll
      for (int t = 0; t < 4; ++t) {
        const int xi = x0 + (t & 1);
        const int yi = y0 + (t >> 1);
        const bool valid = (xi >= 0) & (xi < W) & (yi >= 0) & (yi < H);
        const int  idx = min(max(yi, 0), H - 1) * W + min(max(xi, 0), W - 1);
        const float g = vb[(size_t)idx * D_MODEL];   // coalesced 128B gather
        accf += valid ? (aw * tw[t] * g) : 0.f;
      }
    }
  }
  acc[qf * D_MODEL + h * HEAD_DIM + lane] = accf;
}

// ---------------------------------------------------------------------------
extern "C" void kernel_launch(void* const* d_in, const int* in_sizes, int n_in,
                              void* d_out, int out_size, void* d_ws, size_t ws_size,
                              hipStream_t stream) {
  (void)in_sizes; (void)n_in; (void)out_size; (void)ws_size;

  const float* query  = (const float*)d_in[0];
  const float* refpts = (const float*)d_in[1];
  const float* inflat = (const float*)d_in[2];
  // d_in[3]/d_in[4] (shapes / level starts) are setup constants, hardcoded.
  const float* W_off  = (const float*)d_in[5];
  const float* b_off  = (const float*)d_in[6];
  const float* W_attn = (const float*)d_in[7];
  const float* b_attn = (const float*)d_in[8];
  const float* W_val  = (const float*)d_in[9];
  const float* b_val  = (const float*)d_in[10];
  const float* W_out  = (const float*)d_in[11];
  const float* b_out  = (const float*)d_in[12];
  float* out = (float*)d_out;

  const int M = M_ROWS;
  float* ws      = (float*)d_ws;
  float* v_ws    = ws;                                  // M*256 : value
  float* off_ws  = v_ws    + (size_t)M * D_MODEL;       // M*256 : offsets
  float* attn_ws = off_ws  + (size_t)M * D_MODEL;       // M*128 : attn logits
  float* acc_ws  = attn_ws + (size_t)M * (D_MODEL / 2); // M*256 : acc
  // f16 transposed weights after the fp32 arrays (offset is 16B aligned)
  _Float16* wt_val  = (_Float16*)(acc_ws + (size_t)M * D_MODEL);
  _Float16* wt_off  = wt_val  + 256 * 256;
  _Float16* wt_attn = wt_off  + 256 * 256;
  _Float16* wt_out  = wt_attn + 128 * 256;

  const dim3 blk(256);
  const int  mt32 = (M + 31) / 32;                      // 1263 M-tiles

  // 0) one-time weight transpose + f16 convert
  transpose_w_f16<256><<<dim3(256), blk, 0, stream>>>(W_val,  wt_val);
  transpose_w_f16<256><<<dim3(256), blk, 0, stream>>>(W_off,  wt_off);
  transpose_w_f16<128><<<dim3(128), blk, 0, stream>>>(W_attn, wt_attn);
  transpose_w_f16<256><<<dim3(256), blk, 0, stream>>>(W_out,  wt_out);

  // 1) value projection
  wmma_gemm_bias<256, 4><<<dim3(mt32), blk, 0, stream>>>(inflat, wt_val, b_val, v_ws, M);
  // 2) sampling offsets
  wmma_gemm_bias<256, 4><<<dim3(mt32), blk, 0, stream>>>(query, wt_off, b_off, off_ws, M);
  // 3) attention logits
  wmma_gemm_bias<128, 2><<<dim3(mt32), blk, 0, stream>>>(query, wt_attn, b_attn, attn_ws, M);
  // 4) softmax + deformable sampling
  msda_sample<<<dim3(LEN_TOTAL, NBATCH), blk, 0, stream>>>(refpts, v_ws, off_ws,
                                                           attn_ws, acc_ws);
  // 5) output projection -> d_out
  wmma_gemm_bias<256, 4><<<dim3(mt32), blk, 0, stream>>>(acc_ws, wt_out, b_out, out, M);
}
